// Model_50809463112074
// MI455X (gfx1250) — matrix operand order repaired and checked by reading, not compiled, not run on hardware
//
#include <hip/hip_runtime.h>
#include <hip/hip_bf16.h>

// MeshGraphNets forward on MI455X (gfx1250).
// All dense layers via V_WMMA_F32_16X16X32_BF16 (f32 accum).
// A-tiles are staged once per block into LDS (bank-conflict-free, +8 pad),
// removing the 8x per-wave redundancy; LN fused into final-GEMM epilogue;
// gathers fused into layer-0 staging.

typedef __bf16 bf16;
typedef __attribute__((ext_vector_type(16))) __bf16 bf16x16;
typedef __attribute__((ext_vector_type(8)))  __bf16 bf16x8;
typedef __attribute__((ext_vector_type(8)))  float  f32x8;

#define LAT 128
static const int BB = 2, NN = 8192, EE = 32768, TSTEPS = 15;
static const int BN = BB * NN;   // 16384 node rows
static const int BE = BB * EE;   // 65536 edge rows

#define WMMA_BF16(a, b, c) \
  __builtin_amdgcn_wmma_f32_16x16x32_bf16(false, (a), false, (b), (short)0, (c), false, false)

// ---------------------------------------------------------------------------
// WMMA helpers (fragment layouts per CDNA5 ISA 7.12.2)
// ---------------------------------------------------------------------------

// K-loop with A fragments read from the LDS-staged tile sA[16][K+8].
// A per-lane: row m, chunks [k0+half*8..+7] and [k0+16+half*8..+7].
// B per-lane: W col wave*16+m, rows [k0+half*8..+7] and [k0+16+half*8..+7].
template <int K>
__device__ __forceinline__ f32x8 wmma_from_lds(const bf16* sA,
                                               const bf16* __restrict__ W,
                                               int wave, int lane) {
  const int half = lane >> 4, m = lane & 15;
  const bf16* rp = sA + m * (K + 8);
  f32x8 acc = {};
#pragma unroll
  for (int k0 = 0; k0 < K; k0 += 32) {
    bf16x8 lo = *(const bf16x8*)(rp + k0 + half * 8);
    bf16x8 hi = *(const bf16x8*)(rp + k0 + 16 + half * 8);
    bf16x16 a;
#pragma unroll
    for (int i = 0; i < 8; ++i) { a[i] = lo[i]; a[8 + i] = hi[i]; }
    bf16x16 b;
#pragma unroll
    for (int i = 0; i < 8; ++i) {
      b[i] = W[(size_t)(k0 + half * 8 + i) * LAT + wave * 16 + m];
      b[8 + i] = W[(size_t)(k0 + 16 + half * 8 + i) * LAT + wave * 16 + m];
    }
    acc = WMMA_BF16(a, b, acc);
  }
  return acc;
}

// C/D layout epilogue: element r -> row half*8+r, col wave*16+(lane&15)
template <int RELU, int OUT_BF16>
__device__ __forceinline__ void store_tile(f32x8 acc, const float* __restrict__ bias,
                                           void* __restrict__ out, int lane, int wave,
                                           int tile_m) {
  const int half = lane >> 4, m = lane & 15;
  const int ncol = wave * 16 + m;
  const float bs = bias[ncol];
#pragma unroll
  for (int r = 0; r < 8; ++r) {
    float v = acc[r] + bs;
    if (RELU) v = fmaxf(v, 0.0f);
    const size_t o = (size_t)(tile_m * 16 + half * 8 + r) * LAT + ncol;
    if (OUT_BF16) ((bf16*)out)[o] = (bf16)v;
    else          ((float*)out)[o] = v;
  }
}

// ---------------------------------------------------------------------------
// Generic GEMM: out[M,128] = act(A[M,K] @ W[K,128] + bias). A bf16 row-major.
// grid.x = M/16, block = 256 (8 waves; wave w computes 16-col tile w).
// ---------------------------------------------------------------------------
template <int K, int RELU, int OUT_BF16>
__global__ __launch_bounds__(256) void gemm_k(const bf16* __restrict__ A,
                                              const bf16* __restrict__ W,
                                              const float* __restrict__ bias,
                                              void* __restrict__ out) {
  __shared__ bf16 sA[16 * (K + 8)];
  // cooperative stage: 16 x K bf16 in 8-element chunks
  for (int c = threadIdx.x; c < 16 * (K / 8); c += 256) {
    const int row = c / (K / 8), kc = (c % (K / 8)) * 8;
    *(bf16x8*)(sA + row * (K + 8) + kc) =
        *(const bf16x8*)(A + ((size_t)blockIdx.x * 16 + row) * K + kc);
  }
  __syncthreads();
  const int lane = threadIdx.x & 31, wave = threadIdx.x >> 5;
  f32x8 acc = wmma_from_lds<K>(sA, W, wave, lane);
  store_tile<RELU, OUT_BF16>(acc, bias, out, lane, wave, blockIdx.x);
}

// ---------------------------------------------------------------------------
// GEMM with fused LayerNorm (+ optional residual) epilogue. K = 128.
// Block computes a full 16x128 row-tile, stages result in LDS, each half-wave
// of 16 lanes LN-reduces one row via shuffles.
// ---------------------------------------------------------------------------
template <int HAS_RESID>
__global__ __launch_bounds__(256) void gemm_ln_k(const bf16* __restrict__ A,
                                                 const bf16* __restrict__ W,
                                                 const float* __restrict__ bias,
                                                 const float* __restrict__ g,
                                                 const float* __restrict__ be,
                                                 const float* __restrict__ resid,
                                                 float* __restrict__ out) {
  const int K = LAT;
  __shared__ bf16 sA[16 * (K + 8)];
  __shared__ float tile[16 * 132];
  for (int c = threadIdx.x; c < 16 * (K / 8); c += 256) {
    const int row = c / (K / 8), kc = (c % (K / 8)) * 8;
    *(bf16x8*)(sA + row * (K + 8) + kc) =
        *(const bf16x8*)(A + ((size_t)blockIdx.x * 16 + row) * K + kc);
  }
  __syncthreads();
  const int lane = threadIdx.x & 31, wave = threadIdx.x >> 5;
  const int half = lane >> 4, m = lane & 15;
  f32x8 acc = wmma_from_lds<K>(sA, W, wave, lane);
  const int ncol = wave * 16 + m;
  const float bs = bias[ncol];
#pragma unroll
  for (int r = 0; r < 8; ++r)
    tile[(half * 8 + r) * 132 + ncol] = acc[r] + bs;
  __syncthreads();

  const int lrow = threadIdx.x >> 4, seg = threadIdx.x & 15;
  const float* trow = tile + lrow * 132 + seg * 8;
  float v[8], s = 0.0f, sq = 0.0f;
#pragma unroll
  for (int i = 0; i < 8; ++i) { v[i] = trow[i]; s += v[i]; sq += v[i] * v[i]; }
#pragma unroll
  for (int off = 1; off < 16; off <<= 1) {
    s += __shfl_xor(s, off, 32);
    sq += __shfl_xor(sq, off, 32);
  }
  const float mean = s * (1.0f / LAT);
  const float var = fmaxf(sq * (1.0f / LAT) - mean * mean, 0.0f);
  const float rs = rsqrtf(var + 1e-5f);
  const size_t grow = (size_t)blockIdx.x * 16 + lrow;
#pragma unroll
  for (int i = 0; i < 8; ++i) {
    const int c = seg * 8 + i;
    float o = (v[i] - mean) * rs * g[c] + be[c];
    if (HAS_RESID) o += resid[grow * LAT + c];
    out[grow * LAT + c] = o;
  }
}

// ---------------------------------------------------------------------------
// Edge-MLP layer 0, gather fused into staging:
// A row(e) = [nl[s(e)] | nl[r(e)] | el[e]], K=384, f32 -> bf16 during stage.
// ---------------------------------------------------------------------------
__global__ __launch_bounds__(256) void gemm_edge0_k(const float* __restrict__ nl,
                                                    const float* __restrict__ el,
                                                    const int* __restrict__ snd,
                                                    const int* __restrict__ rcv,
                                                    const bf16* __restrict__ W,
                                                    const float* __restrict__ bias,
                                                    bf16* __restrict__ out) {
  const int K = 3 * LAT;
  __shared__ bf16 sA[16 * (K + 8)];
  __shared__ size_t soff[16], roff[16];
  if (threadIdx.x < 32) {
    const int row = threadIdx.x & 15;
    const int eg = blockIdx.x * 16 + row;
    const int b = eg / EE;
    if (threadIdx.x < 16) soff[row] = ((size_t)b * NN + snd[eg]) * LAT;
    else                  roff[row] = ((size_t)b * NN + rcv[eg]) * LAT;
  }
  __syncthreads();
  // stage 16 x 384 floats (converted) in 8-element chunks: 768 chunks
  for (int c = threadIdx.x; c < 16 * 48; c += 256) {
    const int row = c / 48, k = (c % 48) * 8;
    const float* src;
    if (k < 128)      src = nl + soff[row] + k;
    else if (k < 256) src = nl + roff[row] + (k - 128);
    else              src = el + ((size_t)blockIdx.x * 16 + row) * LAT + (k - 256);
    bf16x8 v;
#pragma unroll
    for (int i = 0; i < 8; ++i) v[i] = (bf16)src[i];
    *(bf16x8*)(sA + row * (K + 8) + k) = v;
  }
  __syncthreads();
  const int lane = threadIdx.x & 31, wave = threadIdx.x >> 5;
  f32x8 acc = wmma_from_lds<K>(sA, W, wave, lane);
  store_tile<1, 1>(acc, bias, out, lane, wave, blockIdx.x);
}

// Node-MLP layer 0, concat fused into staging: A row(n) = [nl[n] | agg[n]], K=256
__global__ __launch_bounds__(256) void gemm_node0_k(const float* __restrict__ nl,
                                                    const float* __restrict__ agg,
                                                    const bf16* __restrict__ W,
                                                    const float* __restrict__ bias,
                                                    bf16* __restrict__ out) {
  const int K = 2 * LAT;
  __shared__ bf16 sA[16 * (K + 8)];
  for (int c = threadIdx.x; c < 16 * 32; c += 256) {
    const int row = c / 32, k = (c % 32) * 8;
    const size_t rb = ((size_t)blockIdx.x * 16 + row) * LAT;
    const float* src = (k < 128) ? nl + rb + k : agg + rb + (k - 128);
    bf16x8 v;
#pragma unroll
    for (int i = 0; i < 8; ++i) v[i] = (bf16)src[i];
    *(bf16x8*)(sA + row * (K + 8) + k) = v;
  }
  __syncthreads();
  const int lane = threadIdx.x & 31, wave = threadIdx.x >> 5;
  f32x8 acc = wmma_from_lds<K>(sA, W, wave, lane);
  store_tile<1, 1>(acc, bias, out, lane, wave, blockIdx.x);
}

// Decoder layer 0: A is f32 (nl), converted during staging. K=128, relu, bf16 out
__global__ __launch_bounds__(256) void gemm_f32a_k(const float* __restrict__ A,
                                                   const bf16* __restrict__ W,
                                                   const float* __restrict__ bias,
                                                   bf16* __restrict__ out) {
  const int K = LAT;
  __shared__ bf16 sA[16 * (K + 8)];
  for (int c = threadIdx.x; c < 16 * 16; c += 256) {
    const int row = c / 16, k = (c % 16) * 8;
    const float* src = A + ((size_t)blockIdx.x * 16 + row) * K + k;
    bf16x8 v;
#pragma unroll
    for (int i = 0; i < 8; ++i) v[i] = (bf16)src[i];
    *(bf16x8*)(sA + row * (K + 8) + k) = v;
  }
  __syncthreads();
  const int lane = threadIdx.x & 31, wave = threadIdx.x >> 5;
  f32x8 acc = wmma_from_lds<K>(sA, W, wave, lane);
  store_tile<1, 1>(acc, bias, out, lane, wave, blockIdx.x);
}

// ---------------------------------------------------------------------------
// Small utility kernels
// ---------------------------------------------------------------------------
__global__ void zero_k(float* __restrict__ p, int n) {
  int i = blockIdx.x * blockDim.x + threadIdx.x;
  if (i < n) p[i] = 0.0f;
}

__global__ void scatter_add_k(const float* __restrict__ el,
                              const int* __restrict__ rcv,
                              float* __restrict__ agg) {
  const size_t tid = (size_t)blockIdx.x * 256 + threadIdx.x;  // BE*128 total
  const int c = (int)(tid & (LAT - 1));
  const size_t e = tid >> 7;
  const int b = (int)(e / EE);
  atomicAdd(&agg[((size_t)b * NN + rcv[e]) * LAT + c], el[tid]);
}

__global__ __launch_bounds__(256) void cvt_pad_k(const float* __restrict__ src,
                                                 bf16* __restrict__ dst,
                                                 int srcRows, int dstRows) {
  const int tid = blockIdx.x * 256 + threadIdx.x;
  if (tid >= dstRows * LAT) return;
  const int row = tid >> 7;
  dst[tid] = (row < srcRows) ? (bf16)src[tid] : (bf16)0.0f;
}

__global__ void node_feat_k(const float* __restrict__ wp,
                            const float* __restrict__ pwp,
                            const int* __restrict__ nt,
                            float* __restrict__ raw) {
  const int i = blockIdx.x * blockDim.x + threadIdx.x;
  if (i >= BN) return;
  float* r = raw + (size_t)i * 12;
  r[0] = wp[i * 3 + 0] - pwp[i * 3 + 0];
  r[1] = wp[i * 3 + 1] - pwp[i * 3 + 1];
  r[2] = wp[i * 3 + 2] - pwp[i * 3 + 2];
  const int t = nt[i];
#pragma unroll
  for (int j = 0; j < 9; ++j) r[3 + j] = (j == t) ? 1.0f : 0.0f;
}

__global__ void edge_feat_k(const float* __restrict__ wp,
                            const float* __restrict__ mp,
                            const int* __restrict__ snd,
                            const int* __restrict__ rcv,
                            float* __restrict__ raw) {
  const int i = blockIdx.x * blockDim.x + threadIdx.x;
  if (i >= BE) return;
  const int b = i / EE;
  const size_t s = (size_t)b * NN + snd[i], r = (size_t)b * NN + rcv[i];
  const float wx = wp[s * 3 + 0] - wp[r * 3 + 0];
  const float wy = wp[s * 3 + 1] - wp[r * 3 + 1];
  const float wz = wp[s * 3 + 2] - wp[r * 3 + 2];
  const float mx = mp[s * 2 + 0] - mp[r * 2 + 0];
  const float my = mp[s * 2 + 1] - mp[r * 2 + 1];
  float* o = raw + (size_t)i * 7;
  o[0] = wx; o[1] = wy; o[2] = wz;
  o[3] = sqrtf(wx * wx + wy * wy + wz * wz);
  o[4] = mx; o[5] = my;
  o[6] = sqrtf(mx * mx + my * my);
}

template <int C>
__global__ __launch_bounds__(256) void stat_partial_k(const float* __restrict__ raw,
                                                      int M, float* __restrict__ part) {
  float s[C] = {}, sq[C] = {};
  for (int row = blockIdx.x * 256 + threadIdx.x; row < M; row += gridDim.x * 256) {
#pragma unroll
    for (int j = 0; j < C; ++j) {
      const float v = raw[(size_t)row * C + j];
      s[j] += v; sq[j] += v * v;
    }
  }
  __shared__ float sh[256][2 * C];
#pragma unroll
  for (int j = 0; j < C; ++j) { sh[threadIdx.x][j] = s[j]; sh[threadIdx.x][C + j] = sq[j]; }
  for (int st = 128; st > 0; st >>= 1) {
    __syncthreads();
    if ((int)threadIdx.x < st)
      for (int j = 0; j < 2 * C; ++j) sh[threadIdx.x][j] += sh[threadIdx.x + st][j];
  }
  if (threadIdx.x == 0)
    for (int j = 0; j < 2 * C; ++j) part[blockIdx.x * 2 * C + j] = sh[0][j];
}

template <int C>
__global__ void stat_final_k(const float* __restrict__ part, int nblk, int M,
                             float* __restrict__ stats) {
  const int j = threadIdx.x;
  if (j >= C) return;
  float s = 0.0f, sq = 0.0f;
  for (int i = 0; i < nblk; ++i) { s += part[i * 2 * C + j]; sq += part[i * 2 * C + C + j]; }
  const float mean = s / M;
  const float var = fmaxf(sq / M - mean * mean, 0.0f);
  stats[2 * j] = mean;
  stats[2 * j + 1] = 1.0f / fmaxf(sqrtf(var), 1e-8f);
}

template <int C>
__global__ void norm_pad_k(const float* __restrict__ raw,
                           const float* __restrict__ stats,
                           bf16* __restrict__ out, int M) {
  const int tid = blockIdx.x * blockDim.x + threadIdx.x;
  if (tid >= M * 32) return;
  const int row = tid >> 5, col = tid & 31;
  float v = 0.0f;
  if (col < C) v = (raw[(size_t)row * C + col] - stats[2 * col]) * stats[2 * col + 1];
  out[tid] = (bf16)v;
}

// Decoder final layer: out[row,0..2] = h[row,:] @ w2[128,3] + b2 (VALU, N=3)
__global__ __launch_bounds__(256) void dec_final_k(const bf16* __restrict__ h,
                                                   const float* __restrict__ w2,
                                                   const float* __restrict__ b2,
                                                   float* __restrict__ out) {
  const int row = (blockIdx.x * 256 + threadIdx.x) >> 5;
  const int lane = threadIdx.x & 31;
  if (row >= BN) return;
  float a0 = 0.0f, a1 = 0.0f, a2 = 0.0f;
  for (int k = lane; k < LAT; k += 32) {
    const float hv = (float)h[(size_t)row * LAT + k];
    a0 += hv * w2[k * 3 + 0];
    a1 += hv * w2[k * 3 + 1];
    a2 += hv * w2[k * 3 + 2];
  }
#pragma unroll
  for (int off = 16; off > 0; off >>= 1) {
    a0 += __shfl_xor(a0, off, 32);
    a1 += __shfl_xor(a1, off, 32);
    a2 += __shfl_xor(a2, off, 32);
  }
  if (lane == 0) {
    out[(size_t)row * 3 + 0] = a0 + b2[0];
    out[(size_t)row * 3 + 1] = a1 + b2[1];
    out[(size_t)row * 3 + 2] = a2 + b2[2];
  }
}

// ---------------------------------------------------------------------------
// Host driver
// ---------------------------------------------------------------------------
extern "C" void kernel_launch(void* const* d_in, const int* in_sizes, int n_in,
                              void* d_out, int out_size, void* d_ws, size_t ws_size,
                              hipStream_t stream) {
  (void)in_sizes; (void)n_in; (void)out_size; (void)ws_size;

  const float* wp  = (const float*)d_in[0];
  const float* pwp = (const float*)d_in[1];
  const float* mp  = (const float*)d_in[2];
  const int*   nt  = (const int*)d_in[3];
  const int*   snd = (const int*)d_in[4];
  const int*   rcv = (const int*)d_in[5];
  const float* en_w0 = (const float*)d_in[6];  const float* en_b0 = (const float*)d_in[7];
  const float* en_w1 = (const float*)d_in[8];  const float* en_b1 = (const float*)d_in[9];
  const float* en_w2 = (const float*)d_in[10]; const float* en_b2 = (const float*)d_in[11];
  const float* en_g  = (const float*)d_in[12]; const float* en_be = (const float*)d_in[13];
  const float* ee_w0 = (const float*)d_in[14]; const float* ee_b0 = (const float*)d_in[15];
  const float* ee_w1 = (const float*)d_in[16]; const float* ee_b1 = (const float*)d_in[17];
  const float* ee_w2 = (const float*)d_in[18]; const float* ee_b2 = (const float*)d_in[19];
  const float* ee_g  = (const float*)d_in[20]; const float* ee_be = (const float*)d_in[21];
  const float* pe_w0 = (const float*)d_in[22]; const float* pe_b0 = (const float*)d_in[23];
  const float* pe_w1 = (const float*)d_in[24]; const float* pe_b1 = (const float*)d_in[25];
  const float* pe_w2 = (const float*)d_in[26]; const float* pe_b2 = (const float*)d_in[27];
  const float* pe_g  = (const float*)d_in[28]; const float* pe_be = (const float*)d_in[29];
  const float* pn_w0 = (const float*)d_in[30]; const float* pn_b0 = (const float*)d_in[31];
  const float* pn_w1 = (const float*)d_in[32]; const float* pn_b1 = (const float*)d_in[33];
  const float* pn_w2 = (const float*)d_in[34]; const float* pn_b2 = (const float*)d_in[35];
  const float* pn_g  = (const float*)d_in[36]; const float* pn_be = (const float*)d_in[37];
  const float* d_w0 = (const float*)d_in[38]; const float* d_b0 = (const float*)d_in[39];
  const float* d_w1 = (const float*)d_in[40]; const float* d_b1 = (const float*)d_in[41];
  const float* d_w2 = (const float*)d_in[42]; const float* d_b2 = (const float*)d_in[43];

  char* base = (char*)d_ws;
  size_t off = 0;
  auto alloc = [&](size_t bytes) { size_t r = off; off = (off + bytes + 255) & ~(size_t)255; return r; };
  float* nl   = (float*)(base + alloc((size_t)BN * LAT * 4));
  float* el   = (float*)(base + alloc((size_t)BE * LAT * 4));
  float* agg  = (float*)(base + alloc((size_t)BN * LAT * 4));
  bf16* hidA  = (bf16*)(base + alloc((size_t)BE * LAT * 2));
  bf16* hidB  = (bf16*)(base + alloc((size_t)BE * LAT * 2));
  bf16* nodeA = (bf16*)(base + alloc((size_t)BN * 32 * 2));
  bf16* edgeA = (bf16*)(base + alloc((size_t)BE * 32 * 2));
  float* raw  = (float*)(base + alloc((size_t)BE * 7 * 4));
  float* part = (float*)(base + alloc((size_t)64 * 24 * 4));
  float* stats = (float*)(base + alloc((size_t)24 * 4));
  bf16* wb_en0 = (bf16*)(base + alloc((size_t)32 * LAT * 2));
  bf16* wb_en1 = (bf16*)(base + alloc((size_t)LAT * LAT * 2));
  bf16* wb_en2 = (bf16*)(base + alloc((size_t)LAT * LAT * 2));
  bf16* wb_ee0 = (bf16*)(base + alloc((size_t)32 * LAT * 2));
  bf16* wb_ee1 = (bf16*)(base + alloc((size_t)LAT * LAT * 2));
  bf16* wb_ee2 = (bf16*)(base + alloc((size_t)LAT * LAT * 2));
  bf16* wb_pe0 = (bf16*)(base + alloc((size_t)TSTEPS * 384 * LAT * 2));
  bf16* wb_pe1 = (bf16*)(base + alloc((size_t)TSTEPS * LAT * LAT * 2));
  bf16* wb_pe2 = (bf16*)(base + alloc((size_t)TSTEPS * LAT * LAT * 2));
  bf16* wb_pn0 = (bf16*)(base + alloc((size_t)TSTEPS * 256 * LAT * 2));
  bf16* wb_pn1 = (bf16*)(base + alloc((size_t)TSTEPS * LAT * LAT * 2));
  bf16* wb_pn2 = (bf16*)(base + alloc((size_t)TSTEPS * LAT * LAT * 2));
  bf16* wb_d0  = (bf16*)(base + alloc((size_t)LAT * LAT * 2));
  bf16* wb_d1  = (bf16*)(base + alloc((size_t)LAT * LAT * 2));

  auto cvt = [&](const float* s, bf16* d, int srcRows, int dstRows) {
    const int total = dstRows * LAT;
    cvt_pad_k<<<(total + 255) / 256, 256, 0, stream>>>(s, d, srcRows, dstRows);
  };
  cvt(en_w0, wb_en0, 12, 32);          cvt(en_w1, wb_en1, LAT, LAT);
  cvt(en_w2, wb_en2, LAT, LAT);
  cvt(ee_w0, wb_ee0, 7, 32);           cvt(ee_w1, wb_ee1, LAT, LAT);
  cvt(ee_w2, wb_ee2, LAT, LAT);
  cvt(pe_w0, wb_pe0, TSTEPS * 384, TSTEPS * 384);
  cvt(pe_w1, wb_pe1, TSTEPS * LAT, TSTEPS * LAT);
  cvt(pe_w2, wb_pe2, TSTEPS * LAT, TSTEPS * LAT);
  cvt(pn_w0, wb_pn0, TSTEPS * 256, TSTEPS * 256);
  cvt(pn_w1, wb_pn1, TSTEPS * LAT, TSTEPS * LAT);
  cvt(pn_w2, wb_pn2, TSTEPS * LAT, TSTEPS * LAT);
  cvt(d_w0, wb_d0, LAT, LAT);          cvt(d_w1, wb_d1, LAT, LAT);

  // node features: build, normalize over (B,N), pad to K=32 bf16
  node_feat_k<<<(BN + 255) / 256, 256, 0, stream>>>(wp, pwp, nt, raw);
  stat_partial_k<12><<<64, 256, 0, stream>>>(raw, BN, part);
  stat_final_k<12><<<1, 32, 0, stream>>>(part, 64, BN, stats);
  norm_pad_k<12><<<(BN * 32 + 255) / 256, 256, 0, stream>>>(raw, stats, nodeA, BN);

  // edge features
  edge_feat_k<<<(BE + 255) / 256, 256, 0, stream>>>(wp, mp, snd, rcv, raw);
  stat_partial_k<7><<<64, 256, 0, stream>>>(raw, BE, part);
  stat_final_k<7><<<1, 32, 0, stream>>>(part, 64, BE, stats);
  norm_pad_k<7><<<(BE * 32 + 255) / 256, 256, 0, stream>>>(raw, stats, edgeA, BE);

  const int gN = BN / 16, gE = BE / 16;

  // encoders
  gemm_k<32, 1, 1><<<gN, 256, 0, stream>>>(nodeA, wb_en0, en_b0, hidA);
  gemm_k<LAT, 1, 1><<<gN, 256, 0, stream>>>(hidA, wb_en1, en_b1, hidB);
  gemm_ln_k<0><<<gN, 256, 0, stream>>>(hidB, wb_en2, en_b2, en_g, en_be, nullptr, nl);

  gemm_k<32, 1, 1><<<gE, 256, 0, stream>>>(edgeA, wb_ee0, ee_b0, hidA);
  gemm_k<LAT, 1, 1><<<gE, 256, 0, stream>>>(hidA, wb_ee1, ee_b1, hidB);
  gemm_ln_k<0><<<gE, 256, 0, stream>>>(hidB, wb_ee2, ee_b2, ee_g, ee_be, nullptr, el);

  // 15 message-passing steps
  for (int t = 0; t < TSTEPS; ++t) {
    const bf16* w0 = wb_pe0 + (size_t)t * 384 * LAT;
    const bf16* w1 = wb_pe1 + (size_t)t * LAT * LAT;
    const bf16* w2 = wb_pe2 + (size_t)t * LAT * LAT;
    gemm_edge0_k<<<gE, 256, 0, stream>>>(nl, el, snd, rcv, w0, pe_b0 + t * LAT, hidA);
    gemm_k<LAT, 1, 1><<<gE, 256, 0, stream>>>(hidA, w1, pe_b1 + t * LAT, hidB);
    gemm_ln_k<1><<<gE, 256, 0, stream>>>(hidB, w2, pe_b2 + t * LAT,
                                         pe_g + t * LAT, pe_be + t * LAT, el, el);

    zero_k<<<(BN * LAT + 255) / 256, 256, 0, stream>>>(agg, BN * LAT);
    scatter_add_k<<<(BE * LAT) / 256, 256, 0, stream>>>(el, rcv, agg);

    const bf16* nw0 = wb_pn0 + (size_t)t * 256 * LAT;
    const bf16* nw1 = wb_pn1 + (size_t)t * LAT * LAT;
    const bf16* nw2 = wb_pn2 + (size_t)t * LAT * LAT;
    gemm_node0_k<<<gN, 256, 0, stream>>>(nl, agg, nw0, pn_b0 + t * LAT, hidA);
    gemm_k<LAT, 1, 1><<<gN, 256, 0, stream>>>(hidA, nw1, pn_b1 + t * LAT, hidB);
    gemm_ln_k<1><<<gN, 256, 0, stream>>>(hidB, nw2, pn_b2 + t * LAT,
                                         pn_g + t * LAT, pn_be + t * LAT, nl, nl);
  }

  // decoder
  gemm_f32a_k<<<gN, 256, 0, stream>>>(nl, wb_d0, d_b0, hidA);
  gemm_k<LAT, 1, 1><<<gN, 256, 0, stream>>>(hidA, wb_d1, d_b1, hidB);
  dec_final_k<<<(BN * 32 + 255) / 256, 256, 0, stream>>>(hidB, d_w2, d_b2, (float*)d_out);
}
